// DifferentiableGMM_45011257262295
// MI455X (gfx1250) — matrix operand order repaired and checked by reading, not compiled, not run on hardware
//
#include <hip/hip_runtime.h>

// ---- MI455X / gfx1250: wave32, WMMA 16x16x32 bf16 -> f32 ----

typedef __attribute__((ext_vector_type(16))) __bf16 v16bf;
typedef __attribute__((ext_vector_type(8)))  __bf16 v8bf;
typedef __attribute__((ext_vector_type(4)))  __bf16 v4bf;
typedef __attribute__((ext_vector_type(8)))  float  v8f;

#define BATCH 32768
#define DIM   512
#define KMIX  512
#define ROWS  32            // output rows per workgroup
#define XS_STRIDE 520       // 512 + 8 bf16 pad -> rotates LDS bank groups per row

// ---------------------------------------------------------------------------
// Prep 1: means f32 -> bf16 (L2-resident operand), plus ||mu_k||^2
// ---------------------------------------------------------------------------
__global__ void gmm_prep_means(const float* __restrict__ means,
                               __bf16* __restrict__ mbf,
                               float* __restrict__ musq) {
  const int row = blockIdx.x;         // one mean row per block
  const int t   = threadIdx.x;        // 128 threads, 4 floats each
  const float4 v = *(const float4*)(means + (size_t)row * DIM + 4 * t);
  v4bf b;
  b[0] = (__bf16)v.x; b[1] = (__bf16)v.y; b[2] = (__bf16)v.z; b[3] = (__bf16)v.w;
  *(v4bf*)(mbf + (size_t)row * DIM + 4 * t) = b;

  float d = v.x * v.x + v.y * v.y + v.z * v.z + v.w * v.w;
  #pragma unroll
  for (int m = 16; m >= 1; m >>= 1) d += __shfl_xor(d, m, 32);

  __shared__ float red[4];
  if ((t & 31) == 0) red[t >> 5] = d;
  __syncthreads();
  if (t == 0) musq[row] = red[0] + red[1] + red[2] + red[3];
}

// ---------------------------------------------------------------------------
// Prep 2: log_softmax(log_weights) and fold per-component constants:
//   alpha_k = -0.5 * exp(-log_var_k)
//   gamma_k = alpha_k*||mu_k||^2 - 0.5*D*log_var_k + lsw_k
// so log_post(b,k) = alpha_k*(||x_b||^2 - 2*cross_bk) + gamma_k
// ---------------------------------------------------------------------------
__global__ void gmm_prep_coeffs(const float* __restrict__ log_weights,
                                const float* __restrict__ log_vars,
                                const float* __restrict__ musq,
                                float* __restrict__ alpha,
                                float* __restrict__ gamma) {
  const int k = threadIdx.x;          // 512 threads = 16 waves
  __shared__ float red[16];
  const float w = log_weights[k];

  float m = w;
  #pragma unroll
  for (int s = 16; s >= 1; s >>= 1) m = fmaxf(m, __shfl_xor(m, s, 32));
  if ((k & 31) == 0) red[k >> 5] = m;
  __syncthreads();
  float gmax = red[0];
  #pragma unroll
  for (int i = 1; i < 16; ++i) gmax = fmaxf(gmax, red[i]);
  __syncthreads();

  float e = __expf(w - gmax);
  float s = e;
  #pragma unroll
  for (int sm = 16; sm >= 1; sm >>= 1) s += __shfl_xor(s, sm, 32);
  if ((k & 31) == 0) red[k >> 5] = s;
  __syncthreads();
  float gsum = 0.0f;
  #pragma unroll
  for (int i = 0; i < 16; ++i) gsum += red[i];

  const float lsw = w - (gmax + __logf(gsum));
  const float lv  = log_vars[k];
  const float a   = -0.5f * __expf(-lv);
  alpha[k] = a;
  gamma[k] = a * musq[k] - 0.5f * (float)DIM * lv + lsw;
}

// ---------------------------------------------------------------------------
// Main fused kernel: 32x512 tile per block; GEMM via bf16 WMMA; fused softmax.
// Waves: m_half = wave>>2 (16 rows), n_chunk = wave&3 (128 cols = 8 N-tiles).
// ---------------------------------------------------------------------------
__global__ void __launch_bounds__(256)
gmm_main(const float* __restrict__ x, const __bf16* __restrict__ mbf,
         const float* __restrict__ alpha, const float* __restrict__ gamma,
         float* __restrict__ out) {
  __shared__ __align__(16) __bf16 xs[ROWS * XS_STRIDE];
  __shared__ float xsq[ROWS];
  __shared__ float pmax[4][ROWS];
  __shared__ float psum[4][ROWS];

  const int tid     = threadIdx.x;
  const int wave    = tid >> 5;
  const int lane    = tid & 31;
  const int lo16    = lane & 15;
  const int hi      = lane >> 4;       // 0: lanes 0-15, 1: lanes 16-31
  const int m_half  = wave >> 2;
  const int n_chunk = wave & 3;
  const int row0    = blockIdx.x * ROWS;

  if (tid < ROWS) xsq[tid] = 0.0f;
  __syncthreads();

  // ---- stage x tile (f32 -> bf16 in LDS) + per-row sum of squares ----
  {
    const int r = tid >> 3;            // 0..31 : row
    const int j = tid & 7;             // 8 threads per row
    const float* xr = x + (size_t)(row0 + r) * DIM;
    __bf16* dst = xs + r * XS_STRIDE;
    float sacc = 0.0f;
    #pragma unroll
    for (int i = 0; i < 16; ++i) {
      const int c = 4 * (j + 8 * i);
      const float4 v = *(const float4*)(xr + c);
      sacc += v.x * v.x + v.y * v.y + v.z * v.z + v.w * v.w;
      v4bf b;
      b[0] = (__bf16)v.x; b[1] = (__bf16)v.y; b[2] = (__bf16)v.z; b[3] = (__bf16)v.w;
      *(v4bf*)(dst + c) = b;
    }
    atomicAdd(&xsq[r], sacc);          // ds_add_f32
  }
  __syncthreads();

  // ---- GEMM: cross = x @ means^T for my 16 rows x 128 cols ----
  v8f acc[8];
  #pragma unroll
  for (int t = 0; t < 8; ++t)
    #pragma unroll
    for (int e = 0; e < 8; ++e) acc[t][e] = 0.0f;

  // A fragment: lane holds row (m_half*16 + lo16); contiguous 16 K values starting at kb + 16*hi (same as B).
  const __bf16* arow = xs + (m_half * 16 + lo16) * XS_STRIDE;
  // B fragment: lane holds mean row n = n_chunk*128 + 16*t + lo16,
  //             contiguous 16 K values starting at kb + 16*hi.
  const __bf16* brow = mbf + (size_t)(n_chunk * 128 + lo16) * DIM + 16 * hi;

  #pragma unroll 2
  for (int kk = 0; kk < DIM / 32; ++kk) {
    const int kb = kk * 32;
    v16bf a;
    *((v8bf*)&a)       = *(const v8bf*)(arow + kb + 16 * hi);       // K {0..7}/{16..23}
    *(((v8bf*)&a) + 1) = *(const v8bf*)(arow + kb + 16 * hi + 8);   // K {8..15}/{24..31}
    #pragma unroll
    for (int t = 0; t < 8; ++t) {
      const __bf16* bp = brow + (size_t)t * 16 * DIM + kb;
      v16bf b;
      *((v8bf*)&b)       = *(const v8bf*)(bp);
      *(((v8bf*)&b) + 1) = *(const v8bf*)(bp + 8);
      acc[t] = __builtin_amdgcn_wmma_f32_16x16x32_bf16(
          false, a, false, b, (short)0, acc[t], false, false);
    }
  }

  // ---- fused epilogue: log_post, then softmax over all 512 columns ----
  // C layout: element j of v8f = row (j + 8*hi) of the 16-row tile, col = lo16.
  float rsq[8];
  #pragma unroll
  for (int j = 0; j < 8; ++j) rsq[j] = xsq[m_half * 16 + 8 * hi + j];

  float al[8], gm[8];
  #pragma unroll
  for (int t = 0; t < 8; ++t) {
    const int n = n_chunk * 128 + 16 * t + lo16;
    al[t] = alpha[n];
    gm[t] = gamma[n];
  }

  float rmax[8];
  #pragma unroll
  for (int j = 0; j < 8; ++j) rmax[j] = -3.402823466e38f;
  #pragma unroll
  for (int t = 0; t < 8; ++t)
    #pragma unroll
    for (int j = 0; j < 8; ++j) {
      const float v = al[t] * (rsq[j] - 2.0f * acc[t][j]) + gm[t];
      acc[t][j] = v;
      rmax[j] = fmaxf(rmax[j], v);
    }

  // max across the 16 lanes sharing each row (masks stay within 16-group)
  #pragma unroll
  for (int j = 0; j < 8; ++j)
    #pragma unroll
    for (int m = 8; m >= 1; m >>= 1)
      rmax[j] = fmaxf(rmax[j], __shfl_xor(rmax[j], m, 32));

  if (lo16 == 0) {
    #pragma unroll
    for (int j = 0; j < 8; ++j)
      pmax[n_chunk][m_half * 16 + 8 * hi + j] = rmax[j];
  }
  __syncthreads();

  float gmax[8], rsum[8];
  #pragma unroll
  for (int j = 0; j < 8; ++j) {
    const int r = m_half * 16 + 8 * hi + j;
    float m = fmaxf(fmaxf(pmax[0][r], pmax[1][r]), fmaxf(pmax[2][r], pmax[3][r]));
    gmax[j] = m;
    rsum[j] = 0.0f;
  }
  #pragma unroll
  for (int t = 0; t < 8; ++t)
    #pragma unroll
    for (int j = 0; j < 8; ++j) {
      const float e = __expf(acc[t][j] - gmax[j]);
      acc[t][j] = e;
      rsum[j] += e;
    }
  #pragma unroll
  for (int j = 0; j < 8; ++j)
    #pragma unroll
    for (int m = 8; m >= 1; m >>= 1)
      rsum[j] += __shfl_xor(rsum[j], m, 32);

  if (lo16 == 0) {
    #pragma unroll
    for (int j = 0; j < 8; ++j)
      psum[n_chunk][m_half * 16 + 8 * hi + j] = rsum[j];
  }
  __syncthreads();

  float rinv[8];
  #pragma unroll
  for (int j = 0; j < 8; ++j) {
    const int r = m_half * 16 + 8 * hi + j;
    rinv[j] = 1.0f / (psum[0][r] + psum[1][r] + psum[2][r] + psum[3][r]);
  }

  #pragma unroll
  for (int t = 0; t < 8; ++t) {
    const int n = n_chunk * 128 + 16 * t + lo16;
    #pragma unroll
    for (int j = 0; j < 8; ++j) {
      const int rg = row0 + m_half * 16 + 8 * hi + j;
      out[(size_t)rg * KMIX + n] = acc[t][j] * rinv[j];
    }
  }
}

// ---------------------------------------------------------------------------
extern "C" void kernel_launch(void* const* d_in, const int* in_sizes, int n_in,
                              void* d_out, int out_size, void* d_ws, size_t ws_size,
                              hipStream_t stream) {
  const float* x           = (const float*)d_in[0];
  const float* means       = (const float*)d_in[1];
  const float* log_vars    = (const float*)d_in[2];
  const float* log_weights = (const float*)d_in[3];
  float* out = (float*)d_out;

  // workspace: [bf16 means 512KB][musq 2KB][alpha 2KB][gamma 2KB]
  __bf16* mbf  = (__bf16*)d_ws;
  float*  musq = (float*)((char*)d_ws + (size_t)KMIX * DIM * sizeof(__bf16));
  float*  alp  = musq + KMIX;
  float*  gam  = alp + KMIX;

  gmm_prep_means <<<KMIX, 128, 0, stream>>>(means, mbf, musq);
  gmm_prep_coeffs<<<1, KMIX, 0, stream>>>(log_weights, log_vars, musq, alp, gam);
  gmm_main       <<<BATCH / ROWS, 256, 0, stream>>>(x, mbf, alp, gam, out);
}
